// MaskedMultiSelfAttention_9706626089330
// MI455X (gfx1250) — compile-verified, matrix operand order checked by reading, not run on hardware
//
#include <hip/hip_runtime.h>
#include <hip/hip_bf16.h>

// ---------------------------------------------------------------------------
// Masked multi-head self-attention, CDNA5 (gfx1250), wave32 + WMMA f16->f32.
//   B=2, T=2048, C=1024, N_HEADS=16, D=64
// Pipeline:
//   0) convert_x : x fp32 -> f16 workspace (enables async A staging in GEMM1)
//   1) qkv_gemm  : qkv = x @ W_qkv + b_qkv -> Q[b][h][t][d], K[b][h][t][d],
//                  V transposed V[b][h][d][t]  (f16 in d_ws)
//   2) attn      : causal flash attention, double-buffered K/V tiles staged
//                  with GLOBAL_LOAD_ASYNC_TO_LDS_B128 (ASYNCcnt), f32 accum
//   3) proj_gemm : out = attn @ W_proj + b_proj (f32 out)
// ---------------------------------------------------------------------------

typedef __attribute__((ext_vector_type(16))) _Float16 v16h;
typedef __attribute__((ext_vector_type(8)))  _Float16 v8h;
typedef __attribute__((ext_vector_type(8)))  float    v8f;

#define WMMA_F32_F16(a, b, c) \
  __builtin_amdgcn_wmma_f32_16x16x32_f16(false, (a), false, (b), (short)0, (c), false, false)

// CDNA5 16-bit A/B fragment layout (ISA 7.12.2): lane holds row (lane&15);
// element e maps to K = kbase + e (e<8) and kbase + 16 + (e-8) (e>=8),
// kbase = 8*(lane>=16).  Two contiguous 16B chunks per lane.
__device__ inline v16h frag_ab(const _Float16* row, int kbase) {
  v8h lo = *(const v8h*)(row + kbase);
  v8h hi = *(const v8h*)(row + kbase + 16);
  return __builtin_shufflevector(lo, hi, 0, 1, 2, 3, 4, 5, 6, 7,
                                          8, 9, 10, 11, 12, 13, 14, 15);
}

// Async DMA: 16 bytes global -> LDS per lane, tracked by ASYNCcnt.
// lds_off = LDS byte offset (low 32 bits of generic shared pointer),
// base    = uniform 64-bit global base (SGPR pair), goff = per-lane byte off.
__device__ inline void async_copy16(unsigned lds_off, const void* base,
                                    unsigned goff) {
  asm volatile("global_load_async_to_lds_b128 %0, %1, %2 offset:0"
               :: "v"(lds_off), "v"(goff), "s"(base) : "memory");
}
__device__ inline unsigned lds_off32(const void* p) {
  return (unsigned)(uintptr_t)p;   // addr[31:0] == LDS byte offset
}
__device__ inline void wait_async0() {
  asm volatile("s_wait_asynccnt 0" ::: "memory");
}

// ------------------------------- constants ---------------------------------
constexpr int TB   = 2;            // batch
constexpr int TT   = 2048;         // seq len
constexpr int TC   = 1024;         // hidden
constexpr int TNH  = 16;           // heads
constexpr int TD   = 64;           // head dim
constexpr int LDA  = 40;           // lds row stride (halves): 32 + 8 pad
constexpr float NEGV   = -1e9f;
constexpr float SCALE2 = 0.125f * 1.44269504088896f;  // 1/sqrt(D) * log2(e)

// =================== kernel 0: x fp32 -> f16 ===============================
__global__ __launch_bounds__(256) void convert_x(const float* __restrict__ x,
                                                 _Float16* __restrict__ xh) {
  size_t i = ((size_t)blockIdx.x * 256 + threadIdx.x) * 8;
  float4 a = *(const float4*)(x + i);
  float4 b = *(const float4*)(x + i + 4);
  v8h h;
  h[0] = (_Float16)a.x; h[1] = (_Float16)a.y; h[2] = (_Float16)a.z; h[3] = (_Float16)a.w;
  h[4] = (_Float16)b.x; h[5] = (_Float16)b.y; h[6] = (_Float16)b.z; h[7] = (_Float16)b.w;
  *(v8h*)(xh + i) = h;
}

// =================== kernel 1: QKV GEMM (x @ W_qkv + b) ====================
// grid (32, 24), 256 threads = 8 waves (4 M x 2 N); wave = 2x4 WMMA tiles.
// A tile staged via async DMA (pure f16 copy); B tile transposed f32->f16.
__global__ __launch_bounds__(256) void qkv_gemm(
    const _Float16* __restrict__ xh, const float* __restrict__ W,
    const float* __restrict__ bias,
    _Float16* __restrict__ q, _Float16* __restrict__ k, _Float16* __restrict__ v)
{
  __shared__ _Float16 ldsA[128 * LDA];
  __shared__ _Float16 ldsB[128 * LDA];
  const int tid  = threadIdx.x;
  const int lane = tid & 31;
  const int wid  = tid >> 5;
  const int half = lane >> 4;
  const int kb   = half * 8;
  const int lm   = lane & 15;
  const int wm   = wid & 3;
  const int wn   = wid >> 2;
  const int m0   = blockIdx.x * 128;
  const int n0   = blockIdx.y * 128;

  v8f acc[2][4] = {};
  for (int k0 = 0; k0 < TC; k0 += 32) {
    __syncthreads();
    // async-stage A tile: xh[m0..+128][k0..+32] (512 x 16B chunks)
#pragma unroll
    for (int i = 0; i < 2; ++i) {
      int c8  = tid + i * 256;
      int row = c8 >> 2;
      int col = (c8 & 3) * 8;
      async_copy16(lds_off32(&ldsA[row * LDA + col]), xh,
                   (unsigned)(((m0 + row) * TC + k0 + col) * sizeof(_Float16)));
    }
    // stage B tile transposed: W[k0+kk][n0+nn] -> ldsB[nn][kk] (fp32->f16)
#pragma unroll
    for (int i = 0; i < 16; ++i) {
      int fl = tid + i * 256;
      int kk = fl >> 7;
      int nn = fl & 127;
      ldsB[nn * LDA + kk] = (_Float16)W[(size_t)(k0 + kk) * (3 * TC) + n0 + nn];
    }
    wait_async0();
    __syncthreads();

    v16h af[2], bf[4];
#pragma unroll
    for (int im = 0; im < 2; ++im)
      af[im] = frag_ab(&ldsA[(wm * 32 + im * 16 + lm) * LDA], kb);
#pragma unroll
    for (int in_ = 0; in_ < 4; ++in_)
      bf[in_] = frag_ab(&ldsB[(wn * 64 + in_ * 16 + lm) * LDA], kb);
#pragma unroll
    for (int im = 0; im < 2; ++im)
#pragma unroll
      for (int in_ = 0; in_ < 4; ++in_)
        acc[im][in_] = WMMA_F32_F16(af[im], bf[in_], acc[im][in_]);
  }

  // scatter-store into head-layout Q/K/V, C-layout: row=vgpr+8*half, col=lane
#pragma unroll
  for (int im = 0; im < 2; ++im) {
#pragma unroll
    for (int in_ = 0; in_ < 4; ++in_) {
      const int col   = n0 + wn * 64 + in_ * 16 + lm;
      const int which = col >> 10;
      const int c     = col & (TC - 1);
      const int head  = c >> 6;
      const int d     = c & (TD - 1);
      const float bv  = bias[col];
#pragma unroll
      for (int r = 0; r < 8; ++r) {
        const int m  = m0 + wm * 32 + im * 16 + r + 8 * half;
        const int b  = m >> 11;              // /T
        const int t  = m & (TT - 1);
        const int bh = b * TNH + head;
        _Float16 hval = (_Float16)(acc[im][in_][r] + bv);
        if (which == 0)      q[((size_t)bh * TT + t) * TD + d] = hval;
        else if (which == 1) k[((size_t)bh * TT + t) * TD + d] = hval;
        else                 v[((size_t)bh * TD + d) * TT + t] = hval;  // transposed
      }
    }
  }
}

// ================= kernel 2: causal flash attention ========================
// grid = B*N_HEADS*(T/128) = 512 blocks, 256 threads = 8 waves.
// Each wave owns 16 query rows. 32-wide K/V tiles are double-buffered and
// staged with async DMA so tile st+1 streams while tile st is computed.
__global__ __launch_bounds__(256) void attn_kernel(
    const _Float16* __restrict__ qg, const _Float16* __restrict__ kg,
    const _Float16* __restrict__ vg, _Float16* __restrict__ og)
{
  constexpr int LDK = 72, LDV = 40, LDP = 40;
  __shared__ _Float16 ldsK[2][32 * LDK];    // K tile [s][d]
  __shared__ _Float16 ldsV[2][64 * LDV];    // V tile [d][s] (pre-transposed)
  __shared__ _Float16 ldsP[8 * 16 * LDP];   // per-wave P bounce buffer

  const int tid  = threadIdx.x;
  const int lane = tid & 31;
  const int wid  = tid >> 5;
  const int half = lane >> 4;
  const int kb   = half * 8;
  const int lm   = lane & 15;

  const int bid  = blockIdx.x;
  const int bh   = bid >> 4;                // b*16 + head
  const int row0 = (bid & 15) * 128;
  const int t0   = row0 + wid * 16;

  const _Float16* qh = qg + (size_t)bh * TT * TD;
  const _Float16* kh = kg + (size_t)bh * TT * TD;
  const _Float16* vh = vg + (size_t)bh * TD * TT;

  // Q fragments for this wave's 16 rows (K-dim = d, two 32-steps)
  v16h qa[2];
  {
    const _Float16* qrow = qh + (size_t)(t0 + lm) * TD;
    qa[0] = frag_ab(qrow, kb);
    qa[1] = frag_ab(qrow + 32, kb);
  }

  v8f o[4] = {};
  float mrow[8], lrow[8];
#pragma unroll
  for (int r = 0; r < 8; ++r) { mrow[r] = -1e30f; lrow[r] = 0.f; }
  _Float16* pw = &ldsP[wid * 16 * LDP];

  const int ntiles  = (row0 + 128) / 32;    // causal extent for this block
  const int lastrow = t0 + 15;

  // async-stage one 32-key K/V tile pair into buffer `buf`
  auto stageKV = [&](int st, int buf) {
    const int sbase = st * 32;
    {   // K tile [32][64]: 256 thr x 16B
      int row = tid >> 3, seg = (tid & 7) * 8;
      async_copy16(lds_off32(&ldsK[buf][row * LDK + seg]), kh,
                   (unsigned)(((sbase + row) * TD + seg) * sizeof(_Float16)));
    }
    {   // V tile [64][32] from transposed V
      int row = tid >> 2, seg = (tid & 3) * 8;
      async_copy16(lds_off32(&ldsV[buf][row * LDV + seg]), vh,
                   (unsigned)((row * TT + sbase + seg) * sizeof(_Float16)));
    }
  };

  stageKV(0, 0);
  for (int st = 0; st < ntiles; ++st) {
    const int cur   = st & 1;
    const int sbase = st * 32;
    wait_async0();              // my DMA for buf[cur] landed in LDS
    __syncthreads();            // everyone's landed; prev compute on buf[cur^1] done
    if (st + 1 < ntiles) stageKV(st + 1, cur ^ 1);   // overlap with compute below
    if (sbase > lastrow) continue;                   // fully masked for this wave

    // ---- scores: S = (Q @ K^T) * scale (log2 domain), two 16x16 tiles
    v8f sc[2];
#pragma unroll
    for (int sub = 0; sub < 2; ++sub) {
      v8f a = {};
      const _Float16* krow = &ldsK[cur][(sub * 16 + lm) * LDK];
      a = WMMA_F32_F16(qa[0], frag_ab(krow, kb), a);
      a = WMMA_F32_F16(qa[1], frag_ab(krow + 32, kb), a);
      sc[sub] = a;
    }

    // ---- mask + online softmax in log2 domain (row = vgpr + 8*half)
    float rmax[8];
#pragma unroll
    for (int r = 0; r < 8; ++r) {
      const int m = t0 + r + 8 * half;
#pragma unroll
      for (int sub = 0; sub < 2; ++sub) {
        const int s = sbase + sub * 16 + lm;
        float vv = sc[sub][r] * SCALE2;
        if (s > m) vv += NEGV;
        sc[sub][r] = vv;
      }
      float rm = fmaxf(sc[0][r], sc[1][r]);
#pragma unroll
      for (int off = 8; off >= 1; off >>= 1)
        rm = fmaxf(rm, __shfl_xor(rm, off, 32));
      rmax[r] = rm;
    }
#pragma unroll
    for (int r = 0; r < 8; ++r) {
      const float mnew  = fmaxf(mrow[r], rmax[r]);
      const float scale = exp2f(mrow[r] - mnew);
      const float p0 = exp2f(sc[0][r] - mnew);
      const float p1 = exp2f(sc[1][r] - mnew);
      sc[0][r] = p0; sc[1][r] = p1;
      float rs = p0 + p1;
#pragma unroll
      for (int off = 8; off >= 1; off >>= 1)
        rs += __shfl_xor(rs, off, 32);
      lrow[r] = lrow[r] * scale + rs;
      mrow[r] = mnew;
#pragma unroll
      for (int dt = 0; dt < 4; ++dt) o[dt][r] *= scale;
    }

    // ---- P (C-layout) -> LDS -> A-layout fragment, then O += P @ V
#pragma unroll
    for (int r = 0; r < 8; ++r) {
      const int mr = r + 8 * half;
      pw[mr * LDP + lm]      = (_Float16)sc[0][r];
      pw[mr * LDP + 16 + lm] = (_Float16)sc[1][r];
    }
    asm volatile("s_wait_dscnt 0" ::: "memory");   // intra-wave cross-lane LDS RAW
    const v16h pf = frag_ab(&pw[lm * LDP], kb);
#pragma unroll
    for (int dt = 0; dt < 4; ++dt)
      o[dt] = WMMA_F32_F16(pf, frag_ab(&ldsV[cur][(dt * 16 + lm) * LDV], kb), o[dt]);
  }

  // ---- epilogue: O /= l, store f16 row-major [B*T][C]
  const int b = bh >> 4, head = bh & 15;
#pragma unroll
  for (int r = 0; r < 8; ++r) {
    const int t = t0 + r + 8 * half;
    const float inv = 1.0f / lrow[r];
#pragma unroll
    for (int dt = 0; dt < 4; ++dt)
      og[((size_t)(b * TT + t)) * TC + head * TD + dt * 16 + lm] =
          (_Float16)(o[dt][r] * inv);
  }
}

// =================== kernel 3: proj GEMM (attn @ W_proj + b) ===============
// grid (32, 8); A (f16) staged via async DMA, B transposed fp32->f16.
__global__ __launch_bounds__(256) void proj_gemm(
    const _Float16* __restrict__ a, const float* __restrict__ W,
    const float* __restrict__ bias, float* __restrict__ out)
{
  __shared__ _Float16 ldsA[128 * LDA];
  __shared__ _Float16 ldsB[128 * LDA];
  const int tid  = threadIdx.x;
  const int lane = tid & 31;
  const int wid  = tid >> 5;
  const int half = lane >> 4;
  const int kb   = half * 8;
  const int lm   = lane & 15;
  const int wm   = wid & 3;
  const int wn   = wid >> 2;
  const int m0   = blockIdx.x * 128;
  const int n0   = blockIdx.y * 128;

  v8f acc[2][4] = {};
  for (int k0 = 0; k0 < TC; k0 += 32) {
    __syncthreads();
#pragma unroll
    for (int i = 0; i < 2; ++i) {           // A tile: 512 x 16B async chunks
      int c8  = tid + i * 256;
      int row = c8 >> 2;
      int col = (c8 & 3) * 8;
      async_copy16(lds_off32(&ldsA[row * LDA + col]), a,
                   (unsigned)(((m0 + row) * TC + k0 + col) * sizeof(_Float16)));
    }
#pragma unroll
    for (int i = 0; i < 16; ++i) {          // B tile transposed fp32->f16
      int fl = tid + i * 256;
      int kk = fl >> 7;
      int nn = fl & 127;
      ldsB[nn * LDA + kk] = (_Float16)W[(size_t)(k0 + kk) * TC + n0 + nn];
    }
    wait_async0();
    __syncthreads();

    v16h af[2], bf[4];
#pragma unroll
    for (int im = 0; im < 2; ++im)
      af[im] = frag_ab(&ldsA[(wm * 32 + im * 16 + lm) * LDA], kb);
#pragma unroll
    for (int in_ = 0; in_ < 4; ++in_)
      bf[in_] = frag_ab(&ldsB[(wn * 64 + in_ * 16 + lm) * LDA], kb);
#pragma unroll
    for (int im = 0; im < 2; ++im)
#pragma unroll
      for (int in_ = 0; in_ < 4; ++in_)
        acc[im][in_] = WMMA_F32_F16(af[im], bf[in_], acc[im][in_]);
  }

#pragma unroll
  for (int im = 0; im < 2; ++im) {
#pragma unroll
    for (int in_ = 0; in_ < 4; ++in_) {
      const int col  = n0 + wn * 64 + in_ * 16 + lm;
      const float bv = bias[col];
#pragma unroll
      for (int r = 0; r < 8; ++r) {
        const int m = m0 + wm * 32 + im * 16 + r + 8 * half;
        out[(size_t)m * TC + col] = acc[im][in_][r] + bv;
      }
    }
  }
}

// ============================== launcher ===================================
extern "C" void kernel_launch(void* const* d_in, const int* in_sizes, int n_in,
                              void* d_out, int out_size, void* d_ws, size_t ws_size,
                              hipStream_t stream) {
  (void)in_sizes; (void)n_in; (void)out_size; (void)ws_size;
  const float* x     = (const float*)d_in[0];
  const float* Wqkv  = (const float*)d_in[1];
  const float* bqkv  = (const float*)d_in[2];
  const float* Wproj = (const float*)d_in[3];
  const float* bproj = (const float*)d_in[4];
  float* out = (float*)d_out;

  const size_t BTC = (size_t)TB * TT * TC;      // 4M elements
  _Float16* ws  = (_Float16*)d_ws;
  _Float16* xh  = ws;
  _Float16* qws = ws + BTC;
  _Float16* kws = ws + 2 * BTC;
  _Float16* vws = ws + 3 * BTC;
  _Float16* aws = ws + 4 * BTC;                 // 40 MB total f16 workspace

  convert_x<<<dim3(BTC / (256 * 8)), 256, 0, stream>>>(x, xh);
  qkv_gemm<<<dim3((TB * TT) / 128, (3 * TC) / 128), 256, 0, stream>>>(
      xh, Wqkv, bqkv, qws, kws, vws);
  attn_kernel<<<dim3(TB * TNH * (TT / 128)), 256, 0, stream>>>(
      qws, kws, vws, aws);
  proj_gemm<<<dim3((TB * TT) / 128, TC / 128), 256, 0, stream>>>(
      aws, Wproj, bproj, out);
}
